// SOKDALayer_62302795596521
// MI455X (gfx1250) — matrix operand order repaired and checked by reading, not compiled, not run on hardware
//
#include <hip/hip_runtime.h>
#include <hip/hip_bf16.h>
#include <math.h>

typedef __attribute__((ext_vector_type(16))) __bf16 v16bf;
typedef __attribute__((ext_vector_type(8)))  __bf16 v8bf;
typedef __attribute__((ext_vector_type(8)))  float  v8f;

#define Bb 8
#define Tt 2048
#define Dd 1024
#define Hh 16
#define DH 64
#define DI 1024

// ---------------- workspace layout (bytes) ----------------
static constexpr size_t OFF_XNB  = 0;                               // 16384*1024*2
static constexpr size_t OFF_WT   = OFF_XNB  + (size_t)16384*1024*2; // 4096*1024*2
static constexpr size_t OFF_WOT  = OFF_WT   + (size_t)4096*1024*2;  // 1024*1024*2
static constexpr size_t OFF_Y    = OFF_WOT  + (size_t)1024*1024*2;  // 16384*4096*4
static constexpr size_t OFF_BETA = OFF_Y    + (size_t)16384*4096*4; // 16384*16*4
static constexpr size_t OFF_YBF  = OFF_BETA + (size_t)16384*16*4;   // 16384*1024*2

// ---------------- LayerNorm -> bf16 ----------------
__global__ __launch_bounds__(256) void ln_kernel(const float* __restrict__ x,
                                                 const float* __restrict__ g,
                                                 const float* __restrict__ be,
                                                 __bf16* __restrict__ xnb) {
  int row = blockIdx.x;                       // B*T rows
  const float* xr = x + (size_t)row * Dd;
  float v[4], s = 0.f, sq = 0.f;
#pragma unroll
  for (int i = 0; i < 4; i++) {
    v[i] = xr[threadIdx.x + i * 256];
    s += v[i]; sq += v[i] * v[i];
  }
  __shared__ float r1[256], r2[256];
  r1[threadIdx.x] = s; r2[threadIdx.x] = sq;
  __syncthreads();
  for (int off = 128; off > 0; off >>= 1) {
    if (threadIdx.x < off) { r1[threadIdx.x] += r1[threadIdx.x + off];
                             r2[threadIdx.x] += r2[threadIdx.x + off]; }
    __syncthreads();
  }
  float mu = r1[0] * (1.f / Dd);
  float var = r2[0] * (1.f / Dd) - mu * mu;
  float rs = rsqrtf(var + 1e-5f);
#pragma unroll
  for (int i = 0; i < 4; i++) {
    int c = threadIdx.x + i * 256;
    xnb[(size_t)row * Dd + c] = (__bf16)((v[i] - mu) * rs * g[c] + be[c]);
  }
}

// ---------------- weight convert: WT[n][k] = W[k][n], concat q|k|v|a ----------------
__global__ __launch_bounds__(256) void wcat_kernel(const float* __restrict__ Wq,
                                                   const float* __restrict__ Wk,
                                                   const float* __restrict__ Wv,
                                                   const float* __restrict__ Wa,
                                                   __bf16* __restrict__ wT) {
  int id = blockIdx.x * 256 + threadIdx.x;    // 4096*1024
  int n = id >> 10, k = id & 1023;
  int sel = n >> 10, col = n & 1023;
  const float* W = (sel == 0) ? Wq : (sel == 1) ? Wk : (sel == 2) ? Wv : Wa;
  wT[id] = (__bf16)W[(size_t)k * DI + col];
}

__global__ __launch_bounds__(256) void wo_kernel(const float* __restrict__ Wo,
                                                 __bf16* __restrict__ woT) {
  int id = blockIdx.x * 256 + threadIdx.x;    // 1024*1024
  int n = id >> 10, k = id & 1023;
  woT[id] = (__bf16)Wo[(size_t)k * Dd + n];
}

// ---------------- beta = sigmoid(xn @ Wb + bb) ----------------
__global__ __launch_bounds__(256) void beta_kernel(const __bf16* __restrict__ xnb,
                                                   const float* __restrict__ Wb,
                                                   const float* __restrict__ bb,
                                                   float* __restrict__ beta) {
  int id = blockIdx.x * 256 + threadIdx.x;    // 16384*16
  int row = id >> 4, h = id & 15;
  const __bf16* xr = xnb + (size_t)row * Dd;
  float acc = bb[h];
#pragma unroll 4
  for (int d = 0; d < Dd; d++) acc = fmaf((float)xr[d], Wb[d * Hh + h], acc);
  beta[id] = 1.f / (1.f + __expf(-acc));
}

// ---------------- bf16 WMMA GEMM: C[M,N] = A[M,K] @ BT[N,K]^T ----------------
// 256 threads / 8 waves, 128x128 block tile, 32x64 wave tile (2x4 WMMA frags),
// double-buffered LDS with register prefetch of the next K-tile.
// mode 0: qkva epilogue (cols >= 3072 -> sigmoid(acc + bias[col-3072]))
// mode 1: residual epilogue (acc + resid)
union FragBf { v16bf v; v8bf h[2]; };

__global__ __launch_bounds__(256) void gemm_kernel(const __bf16* __restrict__ A,
                                                   const __bf16* __restrict__ BT,
                                                   float* __restrict__ C,
                                                   const float* __restrict__ bias,
                                                   const float* __restrict__ resid,
                                                   int M, int N, int K, int mode) {
  __shared__ __bf16 At[2][128 * 40];          // 128 rows x 32 K, stride 40 (80B)
  __shared__ __bf16 Bt[2][128 * 40];
  const int tid = threadIdx.x;
  const int lane = tid & 31, wave = tid >> 5;
  const int r = lane & 15, hh = lane >> 4;
  const int wm = wave >> 1, wn = wave & 1;    // 4x2 wave grid
  const int row0 = blockIdx.y * 128;
  const int col0 = blockIdx.x * 128;
  // staging: 512 chunks of 8 bf16 per tile, 2 per thread
  const int c1 = tid + 256;
  const int ar0 = tid >> 2, ak0 = (tid & 3) * 8;
  const int ar1 = c1 >> 2,  ak1 = (c1 & 3) * 8;

  v8f acc[2][4] = {};
  v8bf ra0, ra1, rb0, rb1;

  // prologue: fetch K-tile 0 into registers
  ra0 = *(const v8bf*)&A [(size_t)(row0 + ar0) * K + ak0];
  ra1 = *(const v8bf*)&A [(size_t)(row0 + ar1) * K + ak1];
  rb0 = *(const v8bf*)&BT[(size_t)(col0 + ar0) * K + ak0];
  rb1 = *(const v8bf*)&BT[(size_t)(col0 + ar1) * K + ak1];

  const int NS = K / 32;
  for (int s = 0; s < NS; ++s) {
    __bf16* Ab = At[s & 1];
    __bf16* Bs = Bt[s & 1];
    *(v8bf*)&Ab[ar0 * 40 + ak0] = ra0;
    *(v8bf*)&Ab[ar1 * 40 + ak1] = ra1;
    *(v8bf*)&Bs[ar0 * 40 + ak0] = rb0;
    *(v8bf*)&Bs[ar1 * 40 + ak1] = rb1;
    if (s + 1 < NS) {                         // prefetch next tile (latency hides under WMMA)
      int k0 = (s + 1) * 32;
      ra0 = *(const v8bf*)&A [(size_t)(row0 + ar0) * K + k0 + ak0];
      ra1 = *(const v8bf*)&A [(size_t)(row0 + ar1) * K + k0 + ak1];
      rb0 = *(const v8bf*)&BT[(size_t)(col0 + ar0) * K + k0 + ak0];
      rb1 = *(const v8bf*)&BT[(size_t)(col0 + ar1) * K + k0 + ak1];
    }
    __syncthreads();
    // A fragments: lanes 0-15 -> K {0..7,16..23}, lanes 16-31 -> K {8..15,24..31}
    FragBf af[2];
#pragma unroll
    for (int mf = 0; mf < 2; mf++) {
      int arow = wm * 32 + mf * 16 + r;
      af[mf].h[0] = *(const v8bf*)&Ab[arow * 40 + hh * 8];
      af[mf].h[1] = *(const v8bf*)&Ab[arow * 40 + 16 + hh * 8];
    }
#pragma unroll
    for (int nb = 0; nb < 4; nb++) {
      // B fragment: N = r (+ tile offset); lanes 0-15 -> K {0..7,16..23}, lanes 16-31 -> K {8..15,24..31}
      FragBf bfv;
      int brow = wn * 64 + nb * 16 + r;
      bfv.h[0] = *(const v8bf*)&Bs[brow * 40 + hh * 8];
      bfv.h[1] = *(const v8bf*)&Bs[brow * 40 + 16 + hh * 8];
#pragma unroll
      for (int mf = 0; mf < 2; mf++)
        acc[mf][nb] = __builtin_amdgcn_wmma_f32_16x16x32_bf16(
            false, af[mf].v, false, bfv.v, (short)0, acc[mf][nb], false, false);
    }
    __syncthreads();
  }

#pragma unroll
  for (int mf = 0; mf < 2; mf++) {
#pragma unroll
    for (int nb = 0; nb < 4; nb++) {
#pragma unroll
      for (int e = 0; e < 8; e++) {           // D layout: M = e + 8*half, N = r
        int grow = row0 + wm * 32 + mf * 16 + e + 8 * hh;
        int gcol = col0 + wn * 64 + nb * 16 + r;
        float val = acc[mf][nb][e];
        if (mode == 0) {
          if (gcol >= 3072) val = 1.f / (1.f + __expf(-(val + bias[gcol - 3072])));
        } else {
          val += resid[(size_t)grow * N + gcol];
        }
        C[(size_t)grow * N + gcol] = val;
      }
    }
  }
}

// ---------------- sequential SOKDA scan: 1 block per (b,h), 64 threads ----------------
__global__ __launch_bounds__(64) void scan_kernel(const float* __restrict__ Y,
                                                  const float* __restrict__ beta,
                                                  const float* __restrict__ raw_gamma,
                                                  __bf16* __restrict__ ybf) {
  const int b = blockIdx.x >> 4;
  const int h = blockIdx.x & 15;
  const int i = threadIdx.x;                  // owns row i of M (regs) and S (LDS)
  __shared__ float S[64 * 65];                // stride 65 -> bank (i+j)%64, conflict-free
  __shared__ float kbuf[64], knb[64], qbuf[64], vbuf[64], kpb[64];
  __shared__ float wr1[2], wr2[2];
  float Mreg[64];
#pragma unroll
  for (int j = 0; j < 64; j++) Mreg[j] = (j == i) ? 1e-6f : 0.f;
#pragma unroll
  for (int j = 0; j < 64; j++) S[i * 65 + j] = 0.f;
  const float gm = 1.f / (1.f + __expf(-raw_gamma[h]));
  __syncthreads();

  for (int t = 0; t < Tt; t++) {
    size_t base = ((size_t)b * Tt + t) * 4096 + h * DH;
    float qv = Y[base + i];
    float kv = Y[base + 1024 + i];
    float vv = Y[base + 2048 + i];
    float av = Y[base + 3072 + i];            // alpha (sigmoid applied in GEMM epilogue)
    float bt = beta[((size_t)b * Tt + t) * Hh + h];
    if (t + 1 < Tt) {                         // prefetch next step's rows into cache
      __builtin_prefetch(&Y[base + 4096 + i], 0, 1);
      __builtin_prefetch(&Y[base + 4096 + 1024 + i], 0, 1);
      __builtin_prefetch(&Y[base + 4096 + 2048 + i], 0, 1);
      __builtin_prefetch(&Y[base + 4096 + 3072 + i], 0, 1);
    }
    kbuf[i] = kv; qbuf[i] = qv; vbuf[i] = vv;

    float p = kv * kv;                        // ||k||^2: shfl reduce + cross-wave combine
#pragma unroll
    for (int m = 16; m >= 1; m >>= 1) p += __shfl_xor(p, m, 32);
    if ((i & 31) == 0) wr1[i >> 5] = p;
    __syncthreads();
    float ss = wr1[0] + wr1[1];
    float kn_i = kv / fmaxf(sqrtf(ss), 1e-12f);
    knb[i] = kn_i;
    __syncthreads();

    float mk = 0.f;                           // M = gm*M + kn kn^T ; Mk = M kn (row i)
#pragma unroll
    for (int j = 0; j < 64; j++) {
      Mreg[j] = fmaf(gm, Mreg[j], kn_i * knb[j]);
      mk = fmaf(Mreg[j], knb[j], mk);
    }
#pragma unroll
    for (int j = 0; j < 64; j++) S[i * 65 + j] *= av;   // S = alpha_row * S

    float p2 = mk * mk;                       // ||Mk||^2 via shfl reduce
#pragma unroll
    for (int m = 16; m >= 1; m >>= 1) p2 += __shfl_xor(p2, m, 32);
    if ((i & 31) == 0) wr2[i >> 5] = p2;
    __syncthreads();
    float s2 = wr2[0] + wr2[1];
    float kt = mk / fmaxf(sqrtf(s2), 1e-6f);  // k_tilde[i]

    float kp = 0.f;                           // kproj[i] = sum_d kn[d] S[d][i] (column dot)
#pragma unroll
    for (int d = 0; d < 64; d++) kp = fmaf(knb[d], S[d * 65 + i], kp);
    kpb[i] = kp;
    __syncthreads();

    float c1 = bt * kt, c2 = bt * kn_i;       // S += b*kn v^T - b*kt kproj^T (row i)
#pragma unroll
    for (int j = 0; j < 64; j++)
      S[i * 65 + j] += c2 * vbuf[j] - c1 * kpb[j];
    __syncthreads();

    float o = 0.f;                            // o[i] = sum_j S[j][i] q[j]
#pragma unroll
    for (int j = 0; j < 64; j++) o = fmaf(S[j * 65 + i], qbuf[j], o);
    ybf[((size_t)b * Tt + t) * DI + h * DH + i] = (__bf16)o;
    __syncthreads();
  }
}

extern "C" void kernel_launch(void* const* d_in, const int* in_sizes, int n_in,
                              void* d_out, int out_size, void* d_ws, size_t ws_size,
                              hipStream_t stream) {
  const float* x    = (const float*)d_in[0];
  const float* ln_g = (const float*)d_in[1];
  const float* ln_b = (const float*)d_in[2];
  const float* Wq   = (const float*)d_in[3];
  const float* Wk   = (const float*)d_in[4];
  const float* Wv   = (const float*)d_in[5];
  const float* Wa   = (const float*)d_in[6];
  const float* ba   = (const float*)d_in[7];
  const float* Wb   = (const float*)d_in[8];
  const float* bb   = (const float*)d_in[9];
  const float* rg   = (const float*)d_in[10];
  const float* Wo   = (const float*)d_in[11];
  float* out = (float*)d_out;
  char* ws = (char*)d_ws;

  __bf16* xnb  = (__bf16*)(ws + OFF_XNB);
  __bf16* wT   = (__bf16*)(ws + OFF_WT);
  __bf16* woT  = (__bf16*)(ws + OFF_WOT);
  float*  Y    = (float*) (ws + OFF_Y);
  float*  betb = (float*) (ws + OFF_BETA);
  __bf16* ybf  = (__bf16*)(ws + OFF_YBF);

  ln_kernel  <<<Bb * Tt, 256, 0, stream>>>(x, ln_g, ln_b, xnb);
  wcat_kernel<<<(4096 * 1024) / 256, 256, 0, stream>>>(Wq, Wk, Wv, Wa, wT);
  wo_kernel  <<<(1024 * 1024) / 256, 256, 0, stream>>>(Wo, woT);
  beta_kernel<<<(Bb * Tt * Hh) / 256, 256, 0, stream>>>(xnb, Wb, bb, betb);

  // fused Q|K|V|alpha projection: 16384 x 4096 x 1024
  gemm_kernel<<<dim3(4096 / 128, (Bb * Tt) / 128), 256, 0, stream>>>(
      xnb, wT, Y, ba, nullptr, Bb * Tt, 4096, 1024, 0);

  scan_kernel<<<Bb * Hh, 64, 0, stream>>>(Y, betb, rg, ybf);

  // output projection + residual: 16384 x 1024 x 1024
  gemm_kernel<<<dim3(1024 / 128, (Bb * Tt) / 128), 256, 0, stream>>>(
      ybf, woT, out, nullptr, x, Bb * Tt, 1024, 1024, 1);
}
